// TransformerPPBlock_63376537420301
// MI455X (gfx1250) — matrix operand order repaired and checked by reading, not compiled, not run on hardware
//
#include <hip/hip_runtime.h>
#include <cstdint>

typedef unsigned short u16;
typedef unsigned int   u32;

typedef __bf16 v16bf __attribute__((ext_vector_type(16)));
typedef float  v8f   __attribute__((ext_vector_type(8)));

static __device__ __forceinline__ float bf2f(u16 h) {
    return __uint_as_float(((u32)h) << 16);
}
static __device__ __forceinline__ u16 f2bf(float f) {
    u32 u = __float_as_uint(f);
    u32 r = u + 0x7FFFu + ((u >> 16) & 1u);
    return (u16)(r >> 16);
}

union FragU {
    uint4 u[2];
    v16bf v;
};

// CDNA5 async memory->LDS DMA (no VGPR staging, tracked by ASYNCcnt).
static __device__ __forceinline__ void async_b128(u32 lds, const void* g) {
    asm volatile("global_load_async_to_lds_b128 %0, %1, off"
                 :: "v"(lds), "v"(g) : "memory");
}
template <int N>
static __device__ __forceinline__ void wait_async() {
    if constexpr (N == 0)      asm volatile("s_wait_asynccnt 0x0" ::: "memory");
    else if constexpr (N == 1) asm volatile("s_wait_asynccnt 0x1" ::: "memory");
    else if constexpr (N == 2) asm volatile("s_wait_asynccnt 0x2" ::: "memory");
    else if constexpr (N == 3) asm volatile("s_wait_asynccnt 0x3" ::: "memory");
    else if constexpr (N == 4) asm volatile("s_wait_asynccnt 0x4" ::: "memory");
    else if constexpr (N == 6) asm volatile("s_wait_asynccnt 0x6" ::: "memory");
    else                       asm volatile("s_wait_asynccnt 0x8" ::: "memory");
}

// ---------------------------------------------------------------------------
// Generic TN GEMM: C[M,N] = alpha * A[M,K] * B[N,K]^T (+resid)(+C), A/B bf16.
// Batched via blockIdx.z: off = (z/zdiv)*O0 + (z%zdiv)*O1 (element offsets).
// Async double-buffered LDS pipeline: tile k+1 is DMA'd while k computes.
// ---------------------------------------------------------------------------
template <int WM_WAVES, int WN_WAVES, int WM_IT, int WN_IT,
          bool OUT_BF16, bool RESID, bool ACCUM>
__global__ __launch_bounds__(WM_WAVES * WN_WAVES * 32)
void gemm_tn_wmma(const u16* __restrict__ A, const u16* __restrict__ B,
                  void* __restrict__ Cv, const float* __restrict__ resid,
                  int K, int lda, int ldb, int ldc,
                  long long aO0, long long aO1, long long bO0, long long bO1,
                  long long cO0, long long cO1, int zdiv, float alpha) {
    constexpr int BM = WM_WAVES * WM_IT * 16;
    constexpr int BN = WN_WAVES * WN_IT * 16;
    constexpr int BK = 32;
    constexpr int NT = WM_WAVES * WN_WAVES * 32;
    constexpr int LS = BK + 8;            // +16B pad: conflict-free, 16B aligned
    constexpr int CHA = (BM * BK / 8) / NT;
    constexpr int CHB = (BN * BK / 8) / NT;
    constexpr int TILE_OPS = CHA + CHB;   // async instrs per wave per tile

    __shared__ u16 sA[2][BM][LS];
    __shared__ u16 sB[2][BN][LS];

    const int tid  = threadIdx.x;
    const int wave = tid >> 5;
    const int lane = tid & 31;
    const int lh   = lane >> 4;           // lane-half selects K sub-block
    const int lm   = lane & 15;           // row-within-tile

    const int z  = blockIdx.z;
    const int z0 = z / zdiv;
    const int z1 = z % zdiv;
    A += (size_t)z0 * aO0 + (size_t)z1 * aO1;
    B += (size_t)z0 * bO0 + (size_t)z1 * bO1;
    const size_t coff = (size_t)z0 * cO0 + (size_t)z1 * cO1;

    const int m0  = blockIdx.y * BM;
    const int n0  = blockIdx.x * BN;
    const int wm0 = (wave % WM_WAVES) * (WM_IT * 16);
    const int wn0 = (wave / WM_WAVES) * (WN_IT * 16);

    // Per-thread DMA source pointers and LDS destinations (both buffers).
    const u16* gA[CHA];
    const u16* gB[CHB];
    u32 lA[2][CHA], lB[2][CHB];
#pragma unroll
    for (int t = 0; t < CHA; ++t) {
        int ci = tid + t * NT;
        int r  = ci >> 2;
        int c  = (ci & 3) * 8;
        gA[t] = A + (size_t)(m0 + r) * lda + c;
        lA[0][t] = (u32)(uintptr_t)&sA[0][r][c];
        lA[1][t] = (u32)(uintptr_t)&sA[1][r][c];
    }
#pragma unroll
    for (int t = 0; t < CHB; ++t) {
        int ci = tid + t * NT;
        int r  = ci >> 2;
        int c  = (ci & 3) * 8;
        gB[t] = B + (size_t)(n0 + r) * ldb + c;
        lB[0][t] = (u32)(uintptr_t)&sB[0][r][c];
        lB[1][t] = (u32)(uintptr_t)&sB[1][r][c];
    }

    v8f acc[WM_IT][WN_IT];
#pragma unroll
    for (int i = 0; i < WM_IT; ++i)
#pragma unroll
        for (int j = 0; j < WN_IT; ++j)
#pragma unroll
            for (int r = 0; r < 8; ++r) acc[i][j][r] = 0.0f;

    auto issueTile = [&](int buf, int kt) {
#pragma unroll
        for (int t = 0; t < CHA; ++t) async_b128(lA[buf][t], gA[t] + kt);
#pragma unroll
        for (int t = 0; t < CHB; ++t) async_b128(lB[buf][t], gB[t] + kt);
    };

    const int nIt = K / BK;
    issueTile(0, 0);

    for (int it = 0; it < nIt; ++it) {
        const int buf = it & 1;
        if (it + 1 < nIt) {
            issueTile(buf ^ 1, (it + 1) * BK);   // prefetch next tile
            wait_async<TILE_OPS>();              // current tile's DMA done
        } else {
            wait_async<0>();
        }
        __syncthreads();

        const u16 (*pA)[LS] = sA[buf];
        const u16 (*pB)[LS] = sB[buf];

        // Fragments per ISA 16-bit layouts:
        // A 16x32: lane holds K chunks {lh*8 .. +7} and {lh*8+16 .. +7}
        // B 32x16: lane holds K chunks {lh*8 .. +7} and {lh*8+16 .. +7}
        FragU fa[WM_IT], fb[WN_IT];
#pragma unroll
        for (int i = 0; i < WM_IT; ++i) {
            int row = wm0 + i * 16 + lm;
            fa[i].u[0] = *(const uint4*)(&pA[row][lh * 8]);
            fa[i].u[1] = *(const uint4*)(&pA[row][lh * 8 + 16]);
        }
#pragma unroll
        for (int j = 0; j < WN_IT; ++j) {
            int row = wn0 + j * 16 + lm;
            fb[j].u[0] = *(const uint4*)(&pB[row][lh * 8]);
            fb[j].u[1] = *(const uint4*)(&pB[row][lh * 8 + 16]);
        }

#pragma unroll
        for (int i = 0; i < WM_IT; ++i)
#pragma unroll
            for (int j = 0; j < WN_IT; ++j)
                acc[i][j] = __builtin_amdgcn_wmma_f32_16x16x32_bf16(
                    false, fa[i].v, false, fb[j].v, (short)0, acc[i][j],
                    false, false);
        __syncthreads();
    }

    // Epilogue: C/D layout — lane (lh,lm): col n0+..+lm, rows +lh*8 + r
#pragma unroll
    for (int i = 0; i < WM_IT; ++i) {
        int mr = m0 + wm0 + i * 16 + lh * 8;
#pragma unroll
        for (int j = 0; j < WN_IT; ++j) {
            int nc = n0 + wn0 + j * 16 + lm;
#pragma unroll
            for (int r = 0; r < 8; ++r) {
                size_t idx = coff + (size_t)(mr + r) * ldc + nc;
                float v = acc[i][j][r] * alpha;
                if constexpr (RESID) v += resid[idx];
                if constexpr (ACCUM) v += ((const float*)Cv)[idx];
                if constexpr (OUT_BF16) ((u16*)Cv)[idx] = f2bf(v);
                else                    ((float*)Cv)[idx] = v;
            }
        }
    }
}

// ---------------------------------------------------------------------------
// Elementwise / helper kernels
// ---------------------------------------------------------------------------
__global__ void cvt_f32_bf16(const float* __restrict__ in, u16* __restrict__ out,
                             size_t n) {
    size_t i = (size_t)blockIdx.x * blockDim.x + threadIdx.x;
    if (i < n) out[i] = f2bf(in[i]);
}

// out[c][r] = in[r][c], converted to bf16. Batched like the GEMM.
template <typename TI>
__global__ void transpose_cvt(const TI* __restrict__ in, u16* __restrict__ out,
                              int ldi, int ldo,
                              long long iO0, long long iO1,
                              long long oO0, long long oO1, int zdiv) {
    __shared__ float tile[32][33];
    int z = blockIdx.z, z0 = z / zdiv, z1 = z % zdiv;
    in  += (size_t)z0 * iO0 + (size_t)z1 * iO1;
    out += (size_t)z0 * oO0 + (size_t)z1 * oO1;

    int c = blockIdx.x * 32 + threadIdx.x;
    int rbase = blockIdx.y * 32;
#pragma unroll
    for (int i = threadIdx.y; i < 32; i += 8) {
        TI v = in[(size_t)(rbase + i) * ldi + c];
        float f;
        if constexpr (sizeof(TI) == 2) f = bf2f((u16)v); else f = (float)v;
        tile[i][threadIdx.x] = f;
    }
    __syncthreads();
    int ocol = rbase + threadIdx.x;
    int obase = blockIdx.x * 32;
#pragma unroll
    for (int i = threadIdx.y; i < 32; i += 8)
        out[(size_t)(obase + i) * ldo + ocol] = f2bf(tile[threadIdx.x][i]);
}

__global__ __launch_bounds__(256)
void rmsnorm_bf16_k(const float* __restrict__ x, const float* __restrict__ w,
                    u16* __restrict__ out, int D, float eps) {
    const size_t row = blockIdx.x;
    const float* xr = x + row * D;
    const int tid = threadIdx.x;
    float ss = 0.f;
    for (int d = tid; d < D; d += 256) { float v = xr[d]; ss += v * v; }
    __shared__ float red[256];
    red[tid] = ss; __syncthreads();
    for (int s = 128; s > 0; s >>= 1) {
        if (tid < s) red[tid] += red[tid + s];
        __syncthreads();
    }
    float scale = rsqrtf(red[0] / (float)D + eps);
    for (int d = tid; d < D; d += 256)
        out[row * D + d] = f2bf(xr[d] * w[d] * scale);
}

__global__ __launch_bounds__(256)
void softmax_rows_bf16(u16* __restrict__ P, int cols) {
    const size_t row = blockIdx.x;
    u16* p = P + row * cols;
    const int tid = threadIdx.x;
    const int nper = cols / 256;  // 4
    float v[8];
    float mx = -3.4e38f;
    for (int i = 0; i < nper; ++i) {
        v[i] = bf2f(p[tid + i * 256]);
        mx = fmaxf(mx, v[i]);
    }
    __shared__ float red[256];
    __shared__ float bval;
    red[tid] = mx; __syncthreads();
    for (int s = 128; s > 0; s >>= 1) {
        if (tid < s) red[tid] = fmaxf(red[tid], red[tid + s]);
        __syncthreads();
    }
    if (tid == 0) bval = red[0];
    __syncthreads();
    float M = bval, sum = 0.f;
    __syncthreads();
    for (int i = 0; i < nper; ++i) { v[i] = __expf(v[i] - M); sum += v[i]; }
    red[tid] = sum; __syncthreads();
    for (int s = 128; s > 0; s >>= 1) {
        if (tid < s) red[tid] += red[tid + s];
        __syncthreads();
    }
    float inv = 1.0f / red[0];
    for (int i = 0; i < nper; ++i) p[tid + i * 256] = f2bf(v[i] * inv);
}

__global__ __launch_bounds__(256)
void gate_top1_k(const u16* __restrict__ xn, const float* __restrict__ gw,
                 const float* __restrict__ gb, const float* __restrict__ grw,
                 float* __restrict__ g, int D) {
    const size_t n = blockIdx.x;
    const int tid = threadIdx.x;
    float acc[8];
#pragma unroll
    for (int e = 0; e < 8; ++e) acc[e] = 0.f;
    for (int d = tid; d < D; d += 256) {
        float xv = bf2f(xn[n * D + d]);
#pragma unroll
        for (int e = 0; e < 8; ++e) acc[e] += xv * gw[(size_t)e * D + d];
    }
    __shared__ float red[256];
    __shared__ float lg[8];
    for (int e = 0; e < 8; ++e) {
        red[tid] = acc[e]; __syncthreads();
        for (int s = 128; s > 0; s >>= 1) {
            if (tid < s) red[tid] += red[tid + s];
            __syncthreads();
        }
        if (tid == 0) lg[e] = red[0] + gb[e];
        __syncthreads();
    }
    if (tid == 0) {
        float ms = 0.f;
        for (int e = 0; e < 8; ++e) ms += lg[e] * lg[e];
        float inv = rsqrtf(ms / 8.0f + 1.1920928955078125e-07f);
        float t[8], tmax = -3.4e38f;
        for (int e = 0; e < 8; ++e) {
            t[e] = (lg[e] * inv * grw[e]) * 2.0f;   // /TEMP, TEMP=0.5
            tmax = fmaxf(tmax, t[e]);
        }
        float sum = 0.f;
        for (int e = 0; e < 8; ++e) sum += __expf(t[e] - tmax);
        float top = 1.0f / sum;                      // softmax value at argmax
        g[n] = top / (top + 1e-6f);
    }
}

__global__ void swiglu_k(const float* __restrict__ h1, u16* __restrict__ act,
                         size_t n) {
    size_t i = (size_t)blockIdx.x * blockDim.x + threadIdx.x;
    if (i >= n) return;
    size_t row = i >> 11;           // FF2 = 2048
    size_t f   = i & 2047;
    float a = h1[row * 4096 + f];
    float b = h1[row * 4096 + 2048 + f];
    float s = b / (1.0f + __expf(-b));  // silu
    act[i] = f2bf(a * s);
}

__global__ void combine_k(const float* __restrict__ x1,
                          const float* __restrict__ comb,
                          const float* __restrict__ g,
                          float* __restrict__ out, size_t n) {
    size_t i = (size_t)blockIdx.x * blockDim.x + threadIdx.x;
    if (i >= n) return;
    out[i] = x1[i] + comb[i] * g[i >> 10];  // D = 1024
}

// ---------------------------------------------------------------------------
extern "C" void kernel_launch(void* const* d_in, const int* in_sizes, int n_in,
                              void* d_out, int out_size, void* d_ws, size_t ws_size,
                              hipStream_t stream) {
    (void)in_sizes; (void)n_in; (void)out_size; (void)ws_size;
    const int Bb = 4, S = 1024, D = 1024, H = 16, dh = 64, E = 8;
    const int FF1 = 4096, FF2 = 2048, N = Bb * S;   // 4096 tokens

    const float* x       = (const float*)d_in[0];
    const float* w_qkv   = (const float*)d_in[1];
    const float* w_out   = (const float*)d_in[2];
    const float* norm1_w = (const float*)d_in[3];
    const float* norm2_w = (const float*)d_in[4];
    const float* gate_w  = (const float*)d_in[5];
    const float* gate_b  = (const float*)d_in[6];
    const float* gate_rw = (const float*)d_in[7];
    const float* w1      = (const float*)d_in[8];
    const float* w2      = (const float*)d_in[9];
    float* out = (float*)d_out;

    size_t off = 0;
    auto alloc = [&](size_t bytes) -> void* {
        void* p = (char*)d_ws + off;
        off += (bytes + 255) & ~(size_t)255;
        return p;
    };
    u16*   w_qkvb = (u16*)  alloc((size_t)3 * D * D * 2);
    u16*   w_outb = (u16*)  alloc((size_t)D * D * 2);
    u16*   w1t    = (u16*)  alloc((size_t)E * FF1 * D * 2);   // [E,FF1,D]
    u16*   w2t    = (u16*)  alloc((size_t)E * D * FF2 * 2);   // [E,D,FF2]
    u16*   xn1    = (u16*)  alloc((size_t)N * D * 2);
    u16*   qkv    = (u16*)  alloc((size_t)N * 3 * D * 2);
    u16*   vt     = (u16*)  alloc((size_t)Bb * H * dh * S * 2);
    u16*   scoreP = (u16*)  alloc((size_t)Bb * H * S * S * 2);
    u16*   o      = (u16*)  alloc((size_t)N * D * 2);
    float* x1     = (float*)alloc((size_t)N * D * 4);
    u16*   xn2    = (u16*)  alloc((size_t)N * D * 2);
    float* g      = (float*)alloc((size_t)N * 4);
    float* h1     = (float*)alloc((size_t)N * FF1 * 4);
    u16*   act    = (u16*)  alloc((size_t)N * FF2 * 2);
    float* comb   = (float*)alloc((size_t)N * D * 4);

    dim3 blk256(256);
    dim3 blkT(32, 8);

    // 1) Weight conversions (TN layout for all GEMMs)
    cvt_f32_bf16<<<(3 * D * D + 255) / 256, blk256, 0, stream>>>(w_qkv, w_qkvb, (size_t)3 * D * D);
    cvt_f32_bf16<<<(D * D + 255) / 256, blk256, 0, stream>>>(w_out, w_outb, (size_t)D * D);
    transpose_cvt<float><<<dim3(FF1 / 32, D / 32, E), blkT, 0, stream>>>(
        w1, w1t, FF1, D, (long long)D * FF1, 0, (long long)FF1 * D, 0, 1);
    transpose_cvt<float><<<dim3(D / 32, FF2 / 32, E), blkT, 0, stream>>>(
        w2, w2t, D, FF2, (long long)FF2 * D, 0, (long long)D * FF2, 0, 1);

    // 2) norm1 + QKV projection
    rmsnorm_bf16_k<<<N, blk256, 0, stream>>>(x, norm1_w, xn1, D, 1e-8f);
    gemm_tn_wmma<2, 4, 4, 2, true, false, false>
        <<<dim3(3 * D / 128, N / 128, 1), blk256, 0, stream>>>(
            xn1, w_qkvb, qkv, nullptr, D, D, D, 3 * D,
            0, 0, 0, 0, 0, 0, 1, 1.0f);

    // 3) attention: transpose V per (b,h); scores; softmax; P*V^T
    transpose_cvt<u16><<<dim3(dh / 32, S / 32, Bb * H), blkT, 0, stream>>>(
        qkv + 2 * D, vt, 3 * D, S,
        (long long)S * 3 * D, (long long)dh,
        (long long)H * dh * S, (long long)dh * S, H);
    gemm_tn_wmma<2, 4, 4, 2, true, false, false>
        <<<dim3(S / 128, S / 128, Bb * H), blk256, 0, stream>>>(
            qkv, qkv + D, scoreP, nullptr, dh, 3 * D, 3 * D, S,
            (long long)S * 3 * D, (long long)dh,
            (long long)S * 3 * D, (long long)dh,
            (long long)H * S * S, (long long)S * S, H, 0.125f);
    softmax_rows_bf16<<<Bb * H * S, blk256, 0, stream>>>(scoreP, S);
    gemm_tn_wmma<2, 4, 4, 1, true, false, false>
        <<<dim3(dh / 64, S / 128, Bb * H), blk256, 0, stream>>>(
            scoreP, vt, o, nullptr, S, S, S, D,
            (long long)H * S * S, (long long)S * S,
            (long long)H * dh * S, (long long)dh * S,
            (long long)S * D, (long long)dh, H, 1.0f);

    // 4) out-proj fused with residual: x1 = x + o @ w_out^T
    gemm_tn_wmma<2, 4, 4, 2, false, true, false>
        <<<dim3(D / 128, N / 128, 1), blk256, 0, stream>>>(
            o, w_outb, x1, x, D, D, D, D, 0, 0, 0, 0, 0, 0, 1, 1.0f);

    // 5) norm2 + gating
    rmsnorm_bf16_k<<<N, blk256, 0, stream>>>(x1, norm2_w, xn2, D, 1e-8f);
    gate_top1_k<<<N, blk256, 0, stream>>>(xn2, gate_w, gate_b, gate_rw, g, D);

    // 6) dense MoE, expert by expert (serialized on stream; h1 scratch reused)
    for (int e = 0; e < E; ++e) {
        gemm_tn_wmma<2, 4, 4, 2, false, false, false>
            <<<dim3(FF1 / 128, N / 128, 1), blk256, 0, stream>>>(
                xn2, w1t + (size_t)e * FF1 * D, h1, nullptr,
                D, D, D, FF1, 0, 0, 0, 0, 0, 0, 1, 1.0f);
        swiglu_k<<<((size_t)N * FF2 + 255) / 256, blk256, 0, stream>>>(
            h1, act, (size_t)N * FF2);
        if (e == 0)
            gemm_tn_wmma<2, 4, 4, 2, false, false, false>
                <<<dim3(D / 128, N / 128, 1), blk256, 0, stream>>>(
                    act, w2t + (size_t)e * D * FF2, comb, nullptr,
                    FF2, FF2, FF2, D, 0, 0, 0, 0, 0, 0, 1, 1.0f);
        else
            gemm_tn_wmma<2, 4, 4, 2, false, false, true>
                <<<dim3(D / 128, N / 128, 1), blk256, 0, stream>>>(
                    act, w2t + (size_t)e * D * FF2, comb, nullptr,
                    FF2, FF2, FF2, D, 0, 0, 0, 0, 0, 0, 1, 1.0f);
    }

    // 7) out = x1 + comb * g
    combine_k<<<((size_t)N * D + 255) / 256, blk256, 0, stream>>>(
        x1, comb, g, out, (size_t)N * D);
}
